// CUTSModel_23441931501890
// MI455X (gfx1250) — matrix operand order repaired and checked by reading, not compiled, not run on hardware
//
#include <hip/hip_runtime.h>

// ---------------------------------------------------------------------------
// MI455X / gfx1250 implementation of the CUTS encoder+decoder forward pass.
// Conv2-4 and both MLP layers run on v_wmma_f32_16x16x32_bf16 (wave32).
// Activations are kept in bf16 NHWC so im2col K (tap-major, cin-minor) is
// contiguous; weights are repacked into fragment-native tiles per launch.
// The MLP GEMMs stage their packed-B strip into LDS with the Tensor Data
// Mover (tensor_load_to_lds + s_wait_tensorcnt), shared by all 8 waves.
// ---------------------------------------------------------------------------

typedef __bf16 bf16_t;
typedef bf16_t v16bf __attribute__((ext_vector_type(16)));
typedef float  v8f   __attribute__((ext_vector_type(8)));
typedef unsigned int v4u __attribute__((ext_vector_type(4)));
typedef int v4i __attribute__((ext_vector_type(4)));
typedef int v8i __attribute__((ext_vector_type(8)));

struct __attribute__((aligned(16))) U4 { unsigned int x, y, z, w; };
union FragBF  { U4 q[2]; v16bf v; };
union OutPack { bf16_t h[8]; U4 q; };

#define B_    8
#define H_    256
#define W_    256
#define HW_   65536
#define PIX_  524288      // B*H*W
#define LAT_  256
#define PD_   147
#define S_    1024
#define NEG_  0.01f
#define EPS_  1e-5f

#if defined(__gfx1250__) && __has_builtin(__builtin_amdgcn_tensor_load_to_lds)
#define HAVE_TDM 1
#else
#define HAVE_TDM 0
#endif

// ------------------------------ conv1 (direct) -----------------------------
__global__ void conv1_direct(const float* __restrict__ x,
                             const float* __restrict__ w1,
                             bf16_t* __restrict__ pre) {
  int idx = blockIdx.x * blockDim.x + threadIdx.x;      // PIX_*32 threads
  if (idx >= PIX_ * 32) return;
  int c = idx & 31;
  int p = idx >> 5;
  int b = p >> 16, hw = p & 65535, h = hw >> 8, w = hw & 255;
  float acc = 0.f;
  for (int cin = 0; cin < 3; ++cin)
    for (int kh = 0; kh < 3; ++kh) {
      int hh = h + kh - 1; if ((unsigned)hh >= H_) continue;
      for (int kw = 0; kw < 3; ++kw) {
        int ww = w + kw - 1; if ((unsigned)ww >= W_) continue;
        acc += x[(((size_t)(b * 3 + cin)) << 16) + (hh << 8) + ww] *
               w1[((c * 3 + cin) * 3 + kh) * 3 + kw];
      }
    }
  pre[(size_t)p * 32 + c] = (bf16_t)acc;
}

// -------------------- WMMA implicit-GEMM conv (layers 2-4) ------------------
// NHWC bf16 activations. K = tap*CIN + cin.  One wave: 16-pixel strip x MPW
// 16-channel output tiles.  A (weights) prepacked per (mt,kt) tile, row-major
// 16x32; B built from two b128 global loads per lane (cin contiguous).
template <int CIN, int COUT, int MPW>
__global__ __launch_bounds__(256) void conv_wmma(
    const bf16_t* __restrict__ act, const bf16_t* __restrict__ apack,
    bf16_t* __restrict__ pre) {
  constexpr int KT  = CIN * 9 / 32;
  constexpr int MT  = COUT / 16;
  constexpr int MTG = MT / MPW;
  constexpr int TPK = CIN / 32;            // K-tiles per 3x3 tap
  const int wave = threadIdx.x >> 5;
  const int lane = threadIdx.x & 31;
  const int col  = lane & 15;              // N (pixel) column
  const int hi   = lane >> 4;              // lane-half (K/M split)
  int unit = blockIdx.x * 8 + wave;
  if (unit >= 32768 * MTG) return;
  int strip = unit / MTG;
  int mtg   = unit - strip * MTG;
  int b  = strip >> 12;
  int rm = strip & 4095;
  int h  = rm >> 4;
  int w0 = (rm & 15) << 4;

  const v8f z8 = {0.f, 0.f, 0.f, 0.f, 0.f, 0.f, 0.f, 0.f};
  v8f acc[MPW];
#pragma unroll
  for (int i = 0; i < MPW; ++i) acc[i] = z8;

  for (int kt = 0; kt < KT; ++kt) {
    int tap  = kt / TPK;
    int cin0 = (kt - tap * TPK) * 32;
    int dh = tap / 3 - 1, dw = tap - (tap / 3) * 3 - 1;
    int hh = h + dh, ww = w0 + col + dw;
    FragBF bf;
    if ((unsigned)hh < H_ && (unsigned)ww < W_) {
      const bf16_t* p = act +
          ((size_t)(((b << 8) + hh) << 8) + ww) * CIN + cin0 + hi * 16;
      bf.q[0] = *(const U4*)p;
      bf.q[1] = *(const U4*)(p + 8);
      __builtin_prefetch(p + (size_t)W_ * CIN, 0, 0);   // global_prefetch_b8
    } else {
      bf.q[0] = U4{0u, 0u, 0u, 0u};
      bf.q[1] = U4{0u, 0u, 0u, 0u};
    }
#pragma unroll
    for (int i = 0; i < MPW; ++i) {
      int mt = mtg * MPW + i;
      const bf16_t* ab = apack + (size_t)(mt * KT + kt) * 512 + col * 32 + hi * 16;
      FragBF af;
      af.q[0] = *(const U4*)ab;
      af.q[1] = *(const U4*)(ab + 8);
      acc[i] = __builtin_amdgcn_wmma_f32_16x16x32_bf16(
          false, af.v, false, bf.v, (short)0, acc[i], false, false);
    }
  }
#pragma unroll
  for (int i = 0; i < MPW; ++i) {
    int mt = mtg * MPW + i;
    int cbase = mt * 16 + hi * 8;
    bf16_t* op = pre +
        ((size_t)(((b << 8) + h) << 8) + (w0 + col)) * COUT + cbase;
    OutPack pk;
#pragma unroll
    for (int r = 0; r < 8; ++r) pk.h[r] = (bf16_t)acc[i][r];
    *(U4*)op = pk.q;
  }
}

// ------------------------------- BatchNorm ----------------------------------
template <int COUT>
__global__ void bn_stats(const bf16_t* __restrict__ pre,
                         float* __restrict__ sum, float* __restrict__ sq) {
  __shared__ float ssum[256], ssq[256];
  int tid = threadIdx.x;
  int c   = tid & (COUT - 1);
  int rep = tid / COUT, reps = 256 / COUT;
  size_t p0 = (size_t)blockIdx.x * 1024;
  float s = 0.f, q = 0.f;
  for (int p = rep; p < 1024; p += reps) {
    float v = (float)pre[(p0 + p) * COUT + c];
    s += v; q += v * v;
  }
  ssum[tid] = s; ssq[tid] = q;
  for (int off = 128; off >= COUT; off >>= 1) {
    __syncthreads();
    if (tid < off) { ssum[tid] += ssum[tid + off]; ssq[tid] += ssq[tid + off]; }
  }
  __syncthreads();
  if (tid < COUT) { atomicAdd(&sum[c], ssum[tid]); atomicAdd(&sq[c], ssq[tid]); }
}

__global__ void bn_finalize(int n, const float* sum, const float* sq,
                            const float* g, const float* bb,
                            float* scale, float* shift) {
  int c = threadIdx.x;
  if (c >= n) return;
  float inv = 1.f / (float)PIX_;
  float m = sum[c] * inv;
  float v = sq[c] * inv - m * m;
  float sc = g[c] * rsqrtf(v + EPS_);
  scale[c] = sc;
  shift[c] = bb[c] - m * sc;
}

template <int COUT, bool WRITE_FEAT>
__global__ void bn_apply(const bf16_t* __restrict__ pre,
                         const float* __restrict__ scale,
                         const float* __restrict__ shift,
                         bf16_t* __restrict__ actout,
                         float* __restrict__ feat) {
  size_t total = (size_t)PIX_ * COUT;
  size_t stride = (size_t)gridDim.x * blockDim.x;
  for (size_t idx = (size_t)blockIdx.x * blockDim.x + threadIdx.x;
       idx < total; idx += stride) {
    int c = (int)(idx & (COUT - 1));
    size_t p = idx / COUT;
    float xv = (float)pre[idx];
    float y = scale[c] * xv + shift[c];
    y = (y >= 0.f) ? y : NEG_ * y;
    actout[idx] = (bf16_t)y;
    if (WRITE_FEAT) {
      size_t b = p >> 16, hw = p & 65535;
      feat[((b << 8) + (size_t)c) * (size_t)HW_ + hw] = y;  // NCHW fp32
    }
  }
}

// ---------------------- gathers / patch extraction --------------------------
__global__ void gather_pts(const bf16_t* __restrict__ act4,
                           const int* __restrict__ hw,
                           float* __restrict__ out, bf16_t* __restrict__ wab) {
  int si = blockIdx.x * 4 + (threadIdx.x >> 6);
  int t  = threadIdx.x & 63;
  if (si >= B_ * S_) return;
  int h = hw[si * 2], w = hw[si * 2 + 1];
  int b = si >> 10;
  const bf16_t* src = act4 + ((size_t)(((b << 8) + h) << 8) + w) * LAT_;
  float* dst = out + (size_t)si * LAT_;
#pragma unroll
  for (int j = 0; j < 4; ++j) {
    int c = t + (j << 6);
    bf16_t v = src[c];
    dst[c] = (float)v;
    if (wab) wab[(size_t)si * LAT_ + c] = v;
  }
}

__global__ void patch_real_k(const float* __restrict__ x,
                             const int* __restrict__ hw,
                             float* __restrict__ out) {
  int idx = blockIdx.x * blockDim.x + threadIdx.x;
  if (idx >= B_ * S_ * 3 * 49) return;
  int bs = idx / 147;
  int r  = idx - bs * 147;
  int c  = r / 49;
  int rr = r - c * 49;
  int ph = rr / 7, pw = rr - ph * 7;
  int b = bs >> 10;
  int h = hw[bs * 2] - 3 + ph;
  int w = hw[bs * 2 + 1] - 3 + pw;
  out[idx] = x[(((size_t)(b * 3 + c)) << 16) + (h << 8) + w];
}

// ------------------------------- MLP GEMM -----------------------------------
// C[M][N] = A[M][lda] * Bp ; A row-major bf16, Bp fragment-packed (per tile:
// 16 columns x 32 K, K contiguous per column).  Unit order is nt-major: all
// 8 waves of a block share one nt strip of Bp (KT*512 bf16 <= 8 KB), which is
// staged into LDS by the Tensor Data Mover (wave 0 issues tensor_load_to_lds,
// waits on TENSORcnt, then the block barriers).  MODE 0: +bias, lrelu -> bf16
// row-major (ld=160).  MODE 1: +bias -> fp32 [M][147].
template <int MODE>
__global__ __launch_bounds__(256) void mlp_gemm(
    const bf16_t* __restrict__ A, int lda, const bf16_t* __restrict__ Bp,
    int KT, int NT, const float* __restrict__ bias, void* __restrict__ outp,
    int M) {
  __shared__ bf16_t ldsb[8 * 512];           // one nt strip (max KT=8)
  const int wave = threadIdx.x >> 5;
  const int lane = threadIdx.x & 31;
  const int col  = lane & 15;
  const int hi   = lane >> 4;
  const int MTT  = M / 16;                   // 512 (divisible by 8) -> nt is
  int unit = blockIdx.x * 8 + wave;          //   uniform across the block
  int nt = unit / MTT;
  int mt = unit - nt * MTT;

  // ---- stage Bp[nt] strip into LDS ----
  const bf16_t* gsrc = Bp + (size_t)nt * KT * 512;
#if HAVE_TDM
  if (wave == 0) {
    unsigned n = (unsigned)(KT * 512);       // elements (2 bytes each)
    unsigned long long ga = (unsigned long long)(size_t)gsrc;
    // D# group 0: count=1, lds_addr=0, global_addr, type=2 ("image")
    v4u g0 = {1u, 0u, (unsigned)(ga & 0xffffffffu),
              (unsigned)((ga >> 32) & 0x1ffffffu) | (2u << 30)};
    // D# group 1: data_size=2B; tensor_dim0=n, tensor_dim1=1,
    //             tile_dim0=n, tile_dim1=1, tensor_dim0_stride=n
    v8i g1 = {(int)(1u << 16),
              (int)((n & 0xffffu) << 16),
              (int)(((n >> 16) & 0xffffu) | (1u << 16)),
              (int)((n & 0xffffu) << 16),
              1,
              (int)n,
              0, 0};
    v4i z4 = {0, 0, 0, 0};
#if defined(__clang_major__) && __clang_major__ >= 23
    v8i z8 = {0, 0, 0, 0, 0, 0, 0, 0};
    __builtin_amdgcn_tensor_load_to_lds(g0, g1, z4, z4, z8, 0);
#else
    __builtin_amdgcn_tensor_load_to_lds(g0, g1, z4, z4, 0);
#endif
    __builtin_amdgcn_s_wait_tensorcnt(0);
  }
  __syncthreads();
#else
  for (int e = threadIdx.x * 8; e < KT * 512; e += 256 * 8)
    *(U4*)(ldsb + e) = *(const U4*)(gsrc + e);
  __syncthreads();
#endif

  v8f acc = {0.f, 0.f, 0.f, 0.f, 0.f, 0.f, 0.f, 0.f};
  for (int kt = 0; kt < KT; ++kt) {
    const bf16_t* ar = A + (size_t)(mt * 16 + col) * lda + kt * 32 + hi * 16;
    FragBF af;
    af.q[0] = *(const U4*)ar;
    af.q[1] = *(const U4*)(ar + 8);
    const bf16_t* br = ldsb + kt * 512 + col * 32 + hi * 16;  // ds_load_b128 x2
    FragBF bf;
    bf.q[0] = *(const U4*)br;
    bf.q[1] = *(const U4*)(br + 8);
    acc = __builtin_amdgcn_wmma_f32_16x16x32_bf16(
        false, af.v, false, bf.v, (short)0, acc, false, false);
  }
  int n  = nt * 16 + col;
  int m0 = mt * 16 + hi * 8;
  if (MODE == 0) {
    float bv = (n < PD_) ? bias[n] : 0.f;
    bf16_t* o = (bf16_t*)outp;
#pragma unroll
    for (int r = 0; r < 8; ++r) {
      float v = acc[r] + bv;
      v = (v >= 0.f) ? v : NEG_ * v;
      o[(size_t)(m0 + r) * 160 + n] = (bf16_t)v;
    }
  } else {
    if (n < PD_) {
      float bv = bias[n];
      float* o = (float*)outp;
#pragma unroll
      for (int r = 0; r < 8; ++r) o[(size_t)(m0 + r) * PD_ + n] = acc[r] + bv;
    }
  }
}

// ------------------------------ weight packing ------------------------------
__global__ void pack_conv(const float* __restrict__ w, bf16_t* __restrict__ apack,
                          int CIN, int COUT) {
  int KT = CIN * 9 / 32;
  int total = COUT * CIN * 9;
  int idx = blockIdx.x * blockDim.x + threadIdx.x;
  if (idx >= total) return;
  int kl   = idx & 31;
  int mrow = (idx >> 5) & 15;
  int kt   = (idx >> 9) % KT;
  int mt   = (idx >> 9) / KT;
  int k = kt * 32 + kl;
  int m = mt * 16 + mrow;
  int tap = k / CIN, cin = k - tap * CIN;
  int kh = tap / 3, kw = tap - kh * 3;
  apack[idx] = (bf16_t)(w[((m * CIN + cin) * 3 + kh) * 3 + kw]);
}

__global__ void pack_mlp(const float* __restrict__ lw, bf16_t* __restrict__ bp,
                         int Pdim, int Ldim, int KT, int NT) {
  int total = NT * KT * 512;
  int idx = blockIdx.x * blockDim.x + threadIdx.x;
  if (idx >= total) return;
  int k  = idx & 31;
  int n  = (idx >> 5) & 15;
  int kt = (idx >> 9) % KT;
  int nt = (idx >> 9) / KT;
  int K = kt * 32 + k;   // contraction index (l or p)
  int N = nt * 16 + n;   // output index (p or q)
  float v = (N < Pdim && K < Ldim) ? lw[(size_t)N * Ldim + K] : 0.f;
  bp[idx] = (bf16_t)v;
}

__global__ void zero_f(float* p, int n) {
  int i = blockIdx.x * blockDim.x + threadIdx.x;
  if (i < n) p[i] = 0.f;
}

// --------------------------------- launch -----------------------------------
extern "C" void kernel_launch(void* const* d_in, const int* in_sizes, int n_in,
                              void* d_out, int out_size, void* d_ws,
                              size_t ws_size, hipStream_t stream) {
  const float* x       = (const float*)d_in[0];
  const int*   anch    = (const int*)d_in[1];
  const int*   posi    = (const int*)d_in[2];
  const float* w1 = (const float*)d_in[3];
  const float* g1 = (const float*)d_in[4];  const float* bb1 = (const float*)d_in[5];
  const float* w2 = (const float*)d_in[6];
  const float* g2 = (const float*)d_in[7];  const float* bb2 = (const float*)d_in[8];
  const float* w3 = (const float*)d_in[9];
  const float* g3 = (const float*)d_in[10]; const float* bb3 = (const float*)d_in[11];
  const float* w4 = (const float*)d_in[12];
  const float* g4 = (const float*)d_in[13]; const float* bb4 = (const float*)d_in[14];
  const float* l1w = (const float*)d_in[15];
  const float* l1b = (const float*)d_in[16];
  const float* l2w = (const float*)d_in[17];
  const float* l2b = (const float*)d_in[18];

  float* feat   = (float*)d_out;                     // [8,256,256,256] NCHW
  float* preal  = feat  + 134217728ull;              // [8,1024,3,7,7]
  float* precon = preal + 1204224ull;                // [8,1024,3,7,7]
  float* wa_out = precon + 1204224ull;               // [8,1024,256]
  float* wp_out = wa_out + 2097152ull;               // [8,1024,256]

  char* ws = (char*)d_ws;
  size_t off = 0;
  auto alloc = [&](size_t bytes) -> char* {
    char* r = ws + off;
    off += (bytes + 255) & ~(size_t)255;
    return r;
  };
  bf16_t* act    = (bf16_t*)alloc((size_t)PIX_ * 256 * 2);  // max-activation
  bf16_t* pre    = (bf16_t*)alloc((size_t)PIX_ * 256 * 2);  // pre-BN
  bf16_t* apack2 = (bf16_t*)alloc((size_t)64  * 288  * 2);
  bf16_t* apack3 = (bf16_t*)alloc((size_t)128 * 576  * 2);
  bf16_t* apack4 = (bf16_t*)alloc((size_t)256 * 1152 * 2);
  bf16_t* b1p    = (bf16_t*)alloc((size_t)10 * 8 * 512 * 2);
  bf16_t* b2p    = (bf16_t*)alloc((size_t)10 * 5 * 512 * 2);
  bf16_t* wab    = (bf16_t*)alloc((size_t)8192 * 256 * 2);
  bf16_t* hbuf   = (bf16_t*)alloc((size_t)8192 * 160 * 2);
  float*  bnsum   = (float*)alloc(480 * 4);
  float*  bnsq    = (float*)alloc(480 * 4);
  float*  bnscale = (float*)alloc(480 * 4);
  float*  bnshift = (float*)alloc(480 * 4);

  dim3 blk(256);
  zero_f<<<2, blk, 0, stream>>>(bnsum, 480);
  zero_f<<<2, blk, 0, stream>>>(bnsq, 480);

  pack_conv<<<(64 * 288 + 255) / 256, blk, 0, stream>>>(w2, apack2, 32, 64);
  pack_conv<<<(128 * 576 + 255) / 256, blk, 0, stream>>>(w3, apack3, 64, 128);
  pack_conv<<<(256 * 1152 + 255) / 256, blk, 0, stream>>>(w4, apack4, 128, 256);
  pack_mlp<<<(10 * 8 * 512) / 256, blk, 0, stream>>>(l1w, b1p, 147, 256, 8, 10);
  pack_mlp<<<(10 * 5 * 512) / 256, blk, 0, stream>>>(l2w, b2p, 147, 147, 5, 10);

  // layer 1 (direct conv, K=27)
  conv1_direct<<<65536, blk, 0, stream>>>(x, w1, pre);
  bn_stats<32><<<512, blk, 0, stream>>>(pre, bnsum + 0, bnsq + 0);
  bn_finalize<<<1, 32, 0, stream>>>(32, bnsum + 0, bnsq + 0, g1, bb1,
                                    bnscale + 0, bnshift + 0);
  bn_apply<32, false><<<4096, blk, 0, stream>>>(pre, bnscale + 0, bnshift + 0,
                                                act, nullptr);
  // layer 2 (WMMA)
  conv_wmma<32, 64, 1><<<16384, blk, 0, stream>>>(act, apack2, pre);
  bn_stats<64><<<512, blk, 0, stream>>>(pre, bnsum + 32, bnsq + 32);
  bn_finalize<<<1, 64, 0, stream>>>(64, bnsum + 32, bnsq + 32, g2, bb2,
                                    bnscale + 32, bnshift + 32);
  bn_apply<64, false><<<4096, blk, 0, stream>>>(pre, bnscale + 32, bnshift + 32,
                                                act, nullptr);
  // layer 3 (WMMA)
  conv_wmma<64, 128, 1><<<32768, blk, 0, stream>>>(act, apack3, pre);
  bn_stats<128><<<512, blk, 0, stream>>>(pre, bnsum + 96, bnsq + 96);
  bn_finalize<<<1, 128, 0, stream>>>(128, bnsum + 96, bnsq + 96, g3, bb3,
                                     bnscale + 96, bnshift + 96);
  bn_apply<128, false><<<4096, blk, 0, stream>>>(pre, bnscale + 96,
                                                 bnshift + 96, act, nullptr);
  // layer 4 (WMMA) + fp32 NCHW feat output
  conv_wmma<128, 256, 2><<<32768, blk, 0, stream>>>(act, apack4, pre);
  bn_stats<256><<<512, blk, 0, stream>>>(pre, bnsum + 224, bnsq + 224);
  bn_finalize<<<1, 256, 0, stream>>>(256, bnsum + 224, bnsq + 224, g4, bb4,
                                     bnscale + 224, bnshift + 224);
  bn_apply<256, true><<<8192, blk, 0, stream>>>(pre, bnscale + 224,
                                                bnshift + 224, act, feat);
  // gathers + patches
  gather_pts<<<2048, blk, 0, stream>>>(act, anch, wa_out, wab);
  gather_pts<<<2048, blk, 0, stream>>>(act, posi, wp_out, nullptr);
  patch_real_k<<<4704, blk, 0, stream>>>(x, anch, preal);
  // MLP decoder (WMMA GEMMs, TDM-staged B)
  mlp_gemm<0><<<640, blk, 0, stream>>>(wab, 256, b1p, 8, 10, l1b, hbuf, 8192);
  mlp_gemm<1><<<640, blk, 0, stream>>>(hbuf, 160, b2p, 5, 10, l2b, precon, 8192);
}
